// UltraFastFoveatedConv2d_7997229105540
// MI455X (gfx1250) — matrix operand order repaired and checked by reading, not compiled, not run on hardware
//
#include <hip/hip_runtime.h>
#include <hip/hip_bf16.h>

typedef __attribute__((ext_vector_type(16))) __bf16     v16bf;
typedef __attribute__((ext_vector_type(8)))  float      v8f;
typedef __attribute__((ext_vector_type(4)))  unsigned   u32x4;
typedef __attribute__((ext_vector_type(4)))  unsigned   uint32x4_t;
typedef __attribute__((ext_vector_type(8)))  int        int32x8_t;
typedef __attribute__((ext_vector_type(4)))  int        int32x4_t;

#define C_IN   32
#define C_OUT  64
#define BATCH  4
#define HH     128
#define WW     128
#define KTAPS  49      // 7x7 effective kernel
#define XP_H   134     // H + 6 halo
#define XP_W   136     // W + 6 halo + alignment pad
#define SLAB_W 134     // dx positions staged per row (covers w+kx-3 for all w,kx)

// workspace layout (bytes)
#define AW_BYTES   (4 * KTAPS * 32 * 16 * 2)   // 200704: 4 M-tiles * 49 taps * 32 lanes * 16 bf16
#define BEFF_OFF   AW_BYTES
#define XPAD_OFF   (AW_BYTES + 256)
#define XPAD_ELEMS (BATCH * XP_H * XP_W * C_IN)   // channel-minor [b][y][x][c]

// ---------------------------------------------------------------------------
// Fold wf through the three depthwise kernels into a 64x32x7x7 effective
// weight, scattered directly into the WMMA 16-bit A-fragment layout
// (contiguous: lane half h holds k 16h..16h+15). K ordering: k = tap*32 + c (tap-major, channel-minor).
// ---------------------------------------------------------------------------
__global__ void fold_weights(const float* __restrict__ wf,
                             const float* __restrict__ w3,
                             const float* __restrict__ w5,
                             const float* __restrict__ w7,
                             __hip_bfloat16* __restrict__ Aw) {
  int tid = blockIdx.x * blockDim.x + threadIdx.x;
  if (tid >= C_OUT * C_IN * KTAPS) return;
  int t = tid % KTAPS;
  int c = (tid / KTAPS) % C_IN;
  int o = tid / (KTAPS * C_IN);
  int ky = t / 7, kx = t % 7;

  float acc = 0.f;
  int dy3 = ky - 2, dx3 = kx - 2;                 // inner 3x3, pad=1
  if (dy3 >= 0 && dy3 < 3 && dx3 >= 0 && dx3 < 3) {
    for (int k = 0; k < 9; ++k)
      acc += wf[o * 1568 + c * 9 + k] * w3[(c * 9 + k) * 9 + dy3 * 3 + dx3];
  }
  int dy5 = ky - 1, dx5 = kx - 1;                 // ring 5x5, pad=2
  if (dy5 >= 0 && dy5 < 5 && dx5 >= 0 && dx5 < 5) {
    for (int k = 0; k < 16; ++k)
      acc += wf[o * 1568 + 288 + c * 16 + k] * w5[(c * 16 + k) * 25 + dy5 * 5 + dx5];
  }
  for (int k = 0; k < 24; ++k)                    // ring 7x7, pad=3
    acc += wf[o * 1568 + 800 + c * 24 + k] * w7[(c * 24 + k) * 49 + t];

  // A-fragment slot: lane grp0 holds K {0..15}, grp1 {16..31}
  int kgroup   = c >> 4;
  int cc       = c & 15;
  int v        = cc >> 1;
  int half_sel = c & 1;
  int lane     = (o & 15) + 16 * kgroup;
  int mt       = o >> 4;
  Aw[(((mt * KTAPS + t) * 32 + lane) * 8 + v) * 2 + half_sel] = __float2bfloat16(acc);
}

// ---------------------------------------------------------------------------
// beff[o] = bf[o] + wf[o,:] . concat(b3,b5,b7)
// ---------------------------------------------------------------------------
__global__ void fold_bias(const float* __restrict__ wf,
                          const float* __restrict__ b3,
                          const float* __restrict__ b5,
                          const float* __restrict__ b7,
                          const float* __restrict__ bfin,
                          float* __restrict__ beff) {
  int o = threadIdx.x;
  if (o >= C_OUT) return;
  float acc = bfin[o];
  for (int j = 0; j < 288; ++j) acc += wf[o * 1568 + j]       * b3[j];
  for (int j = 0; j < 512; ++j) acc += wf[o * 1568 + 288 + j] * b5[j];
  for (int j = 0; j < 768; ++j) acc += wf[o * 1568 + 800 + j] * b7[j];
  beff[o] = acc;
}

// ---------------------------------------------------------------------------
// Zero-padded, channel-minor bf16 input: xp[b][y][x][c] = x[b][c][y-3][x-3]
// Channel-minor makes the per-block slab a contiguous-row 2D tile -> TDM.
// ---------------------------------------------------------------------------
__global__ void pad_input_cm(const float* __restrict__ x,
                             __hip_bfloat16* __restrict__ xp) {
  int idx = blockIdx.x * blockDim.x + threadIdx.x;
  if (idx >= XPAD_ELEMS) return;
  int c  = idx & 31;
  int xc = (idx >> 5) % XP_W;
  int y  = (idx / (32 * XP_W)) % XP_H;
  int b  = idx / (32 * XP_W * XP_H);
  float v = 0.f;
  int sy = y - 3, sx = xc - 3;
  if (sy >= 0 && sy < HH && sx >= 0 && sx < WW)
    v = x[(((size_t)b * C_IN + c) * HH + sy) * WW + sx];
  xp[idx] = __float2bfloat16(v);
}

// ---------------------------------------------------------------------------
// Implicit-GEMM dense 7x7 conv via WMMA bf16 + TDM slab staging.
// Block = (b, h): full 128-wide output row. 8 waves = 4 M-tiles x 2 N-waves;
// each wave: 16(M) x 64(N) = 4 accumulators, A fragment reused 4x per tap.
// Slab (7 x 134 x 32ch bf16 = 60KB LDS) staged by one tensor_load_to_lds.
// ---------------------------------------------------------------------------
__launch_bounds__(256)
__global__ void fovconv_wmma(const __hip_bfloat16* __restrict__ Aw,
                             const float* __restrict__ beff,
                             const __hip_bfloat16* __restrict__ xpad,
                             float* __restrict__ out) {
  // channel-minor slab: [dy 0..6][dx 0..133][c 0..31]
  __shared__ __align__(32) __hip_bfloat16 slab[7 * SLAB_W * 32];

  int blk = blockIdx.x;                 // BATCH * HH = 512
  int h   = blk & (HH - 1);
  int b   = blk >> 7;

  int tid  = threadIdx.x;
  int lane = tid & 31;
  int wv   = tid >> 5;                  // 0..7

  // ---- TDM: DMA the 2D tile (7 rows x 4288 contiguous elems) into LDS
  if (tid < 32) {
    unsigned long long gaddr = (unsigned long long)(const void*)(
        xpad + ((size_t)(b * XP_H + h) * XP_W) * 32);
    unsigned lds_addr = (unsigned)(unsigned long long)(const void*)&slab[0];

    uint32x4_t g0;
    g0[0] = 1u;                                        // count=1, user D#
    g0[1] = lds_addr;                                  // LDS byte address
    g0[2] = (unsigned)(gaddr & 0xFFFFFFFFull);         // global_addr[31:0]
    g0[3] = (unsigned)((gaddr >> 32) & 0x01FFFFFFull)  // global_addr[56:32]
            | (2u << 30);                              // type = 2 (image)

    int32x8_t g1;
    g1[0] = (1 << 16);            // wg_mask=0, data_size=1 (2B), no extras
    g1[1] = (int)(0xFFFFu << 16); // tensor_dim0[15:0] = 0xFFFF (no OOB clip)
    g1[2] = (int)(0xFFFFu << 16); // dim0[31:16]=0, tensor_dim1[15:0]=0xFFFF
    g1[3] = (SLAB_W * 32) << 16;  // dim1[31:16]=0, tile_dim0 = 4288
    g1[4] = 7;                    // tile_dim1 = 7 rows, tile_dim2 = 0
    g1[5] = XP_W * 32;            // tensor_dim0_stride = 4352 elements
    g1[6] = 0;                    // stride hi, tensor_dim1_stride lo
    g1[7] = 0;
    int32x4_t gz  = {0, 0, 0, 0};              // groups 2/3 unused (2D tile)
    int32x8_t gz8 = {0, 0, 0, 0, 0, 0, 0, 0};  // extra descriptor group: unused

    __builtin_amdgcn_tensor_load_to_lds(g0, g1, gz, gz, gz8, 0);
    __builtin_amdgcn_s_wait_tensorcnt(0);
  }
  __syncthreads();

  int mt     = wv & 3;                  // output channels mt*16 .. mt*16+15
  int nt     = wv >> 2;                 // N offset nt*64
  int kgroup = lane >> 4;
  int ncol   = lane & 15;
  int nwoff  = nt * 64;

  v8f acc[4] = {{}, {}, {}, {}};
  const u32x4* awq = (const u32x4*)Aw;

  for (int ky = 0; ky < 7; ++ky) {
#pragma unroll
    for (int kx = 0; kx < 7; ++kx) {
      int t = ky * 7 + kx;
      // A fragment: 32B contiguous per lane, prearranged layout
      union { u32x4 q[2]; v16bf v; } a;
      const u32x4* ap = awq + (size_t)((mt * KTAPS + t) * 32 + lane) * 2;
      a.q[0] = ap[0];
      a.q[1] = ap[1];
      // 4 B fragments; lane kgroup holds channels kgroup*16..+15 contiguous
      int base = (ky * SLAB_W + nwoff + ncol + kx) * 32 + kgroup * 16;
#pragma unroll
      for (int s = 0; s < 4; ++s) {
        union { u32x4 q[2]; v16bf v; } bf;
        const u32x4* bp = (const u32x4*)&slab[base + s * (16 * 32)];
        bf.q[0] = bp[0];
        bf.q[1] = bp[1];
        acc[s] = __builtin_amdgcn_wmma_f32_16x16x32_bf16(
            false, a.v, false, bf.v, (short)0, acc[s], false, false);
      }
    }
  }

  // D layout: VGPR r, lanes 0-15 -> M=r, N=lane; lanes 16-31 -> M=r+8
  int mhalf = 8 * kgroup;
#pragma unroll
  for (int r = 0; r < 8; ++r) {
    int o = mt * 16 + r + mhalf;
    float bias = beff[o];
    size_t row = (((size_t)b * C_OUT + o) * HH + h) * WW;
#pragma unroll
    for (int s = 0; s < 4; ++s)
      out[row + nwoff + s * 16 + ncol] = acc[s][r] + bias;
  }
}

extern "C" void kernel_launch(void* const* d_in, const int* in_sizes, int n_in,
                              void* d_out, int out_size, void* d_ws, size_t ws_size,
                              hipStream_t stream) {
  const float* x  = (const float*)d_in[0];
  const float* w3 = (const float*)d_in[1];
  const float* b3 = (const float*)d_in[2];
  const float* w5 = (const float*)d_in[3];
  const float* b5 = (const float*)d_in[4];
  const float* w7 = (const float*)d_in[5];
  const float* b7 = (const float*)d_in[6];
  const float* wf = (const float*)d_in[7];
  const float* bf = (const float*)d_in[8];
  float* outp = (float*)d_out;

  __hip_bfloat16* Aw   = (__hip_bfloat16*)d_ws;
  float*          beff = (float*)((char*)d_ws + BEFF_OFF);
  __hip_bfloat16* xpad = (__hip_bfloat16*)((char*)d_ws + XPAD_OFF);

  int wthreads = C_OUT * C_IN * KTAPS;                      // 100352
  fold_weights<<<(wthreads + 255) / 256, 256, 0, stream>>>(wf, w3, w5, w7, Aw);
  fold_bias<<<1, 64, 0, stream>>>(wf, b3, b5, b7, bf, beff);
  pad_input_cm<<<(XPAD_ELEMS + 255) / 256, 256, 0, stream>>>(x, xpad);
  fovconv_wmma<<<BATCH * HH, 256, 0, stream>>>(Aw, beff, xpad, outp);
}
